// TemporalMoEBlock_88484916232436
// MI455X (gfx1250) — matrix operand order repaired and checked by reading, not compiled, not run on hardware
//
#include <hip/hip_runtime.h>
#include <hip/hip_bf16.h>
#include <math.h>

// ---------------- problem constants (from reference) ----------------
constexpr int Bb   = 4;
constexpr int Ss   = 1024;
constexpr int Dd   = 1024;
constexpr int Hh   = 16;
constexpr int Ee   = 8;
constexpr int Kt   = 2;          // top-k
constexpr int HIDd = 4096;
constexpr int HDd  = 64;         // head dim
constexpr int NT   = Bb * Ss;    // 4096 tokens
constexpr int NAS  = NT * Kt;    // 8192 assignment rows

typedef __attribute__((ext_vector_type(16))) _Float16 v16h;
typedef __attribute__((ext_vector_type(8)))  float    v8f;

union F16Frag { v16h v; _Float16 h[16]; uint4 q[2]; };

static __device__ __forceinline__ v8f wmma_f16(const v16h& a, const v16h& b, const v8f& c) {
    return __builtin_amdgcn_wmma_f32_16x16x32_f16(false, a, false, b, (short)0, c, false, false);
}

// ---------------- CDNA5 async / TDM feature detection ----------------
#if defined(__has_builtin)
#if __has_builtin(__builtin_amdgcn_global_load_async_to_lds_b128) && __has_builtin(__builtin_amdgcn_s_wait_asynccnt)
#define USE_ASYNC_LDS 1
#endif
#if __has_builtin(__builtin_amdgcn_tensor_load_to_lds) && __has_builtin(__builtin_amdgcn_s_wait_tensorcnt)
#define USE_TDM 1
#endif
#endif

#ifdef USE_ASYNC_LDS
typedef __attribute__((ext_vector_type(4))) int i4_t;
typedef __attribute__((address_space(1))) i4_t gi4_t;   // global v4i
typedef __attribute__((address_space(3))) i4_t li4_t;   // LDS v4i
// per-lane 16B async copy global->LDS (tracked by ASYNCcnt).
// LDS generic addresses truncate to the 32-bit LDS offset (ISA flat addressing).
static __device__ __forceinline__ void async_cp16(const void* g, void* l) {
    __builtin_amdgcn_global_load_async_to_lds_b128(
        (gi4_t*)(size_t)g,
        (li4_t*)(unsigned)(size_t)l, 0, 0);
}
#endif

#ifdef USE_TDM
typedef __attribute__((ext_vector_type(4))) unsigned tdm_u32x4;
typedef __attribute__((ext_vector_type(8))) int      tdm_i32x8;
typedef __attribute__((ext_vector_type(4))) int      tdm_i32x4;
// 2D f16 tile DMA: tile_x elements per row, tile_y rows, row stride stride_elems.
// D# packing per CDNA5 ISA (group0 128b, group1 256b).
static __device__ __forceinline__ void tdm_load_tile_f16(
        void* lds_dst, const void* gsrc, int tile_x, int tile_y,
        unsigned td0, unsigned td1, unsigned long long stride_elems) {
    unsigned long long ga = (unsigned long long)(size_t)gsrc;
    unsigned lds_off = (unsigned)(size_t)lds_dst;
    tdm_u32x4 g0;
    g0.x = 1u;                                                  // count=1 (valid), user mode
    g0.y = lds_off;                                             // lds_addr
    g0.z = (unsigned)ga;                                        // global_addr[31:0]
    g0.w = (unsigned)((ga >> 32) & 0x01ffffffu) | (2u << 30);   // global_addr[56:32] | type=2
    tdm_i32x8 g1;
    g1[0] = 0x10000;                                            // workgroup_mask=0, data_size=1 (2B)
    g1[1] = (int)((td0 & 0xffffu) << 16);                       // tensor_dim0[15:0] @ bits63:48
    g1[2] = (int)((td0 >> 16) | ((td1 & 0xffffu) << 16));       // td0[31:16] | td1[15:0]
    g1[3] = (int)((td1 >> 16) | ((unsigned)(tile_x & 0xffff) << 16)); // td1[31:16] | tile_dim0
    g1[4] = (int)(tile_y & 0xffff);                             // tile_dim1 (tile_dim2=0)
    g1[5] = (int)(unsigned)(stride_elems & 0xffffffffu);        // tensor_dim0_stride[31:0]
    g1[6] = (int)(unsigned)((stride_elems >> 32) & 0xffffu);    // stride[47:32] (dim1_stride=0)
    g1[7] = 0;
    tdm_i32x4 gz4 = {0, 0, 0, 0};
    tdm_i32x8 gz8 = {0, 0, 0, 0, 0, 0, 0, 0};
    __builtin_amdgcn_tensor_load_to_lds(g0, g1, gz4, gz4, gz8, 0);
}
#endif

// ---------------- weight convert + transpose: f32 [K,N] -> f16 [N,K] ----------------
__global__ void cvt_transpose_kernel(const float* __restrict__ src, _Float16* __restrict__ dst,
                                     int Kd, int Nd) {
    size_t base = (size_t)blockIdx.z * Kd * Nd;
    int idx = blockIdx.x * 256 + threadIdx.x;
    if (idx < Kd * Nd) {
        int k = idx / Nd, n = idx % Nd;
        dst[base + (size_t)n * Kd + k] = (_Float16)src[base + idx];
    }
}

// ---------------- LayerNorm (f32 row -> f16 row) ----------------
__global__ __launch_bounds__(256) void ln_f16_kernel(const float* __restrict__ x,
                                                     const float* __restrict__ g,
                                                     const float* __restrict__ b,
                                                     _Float16* __restrict__ out) {
    int row = blockIdx.x;
    const float* xr = x + (size_t)row * Dd;
    float s = 0.f, s2 = 0.f;
    for (int i = threadIdx.x; i < Dd; i += 256) { float v = xr[i]; s += v; s2 += v * v; }
    __shared__ float red[2][8];
    for (int o = 16; o > 0; o >>= 1) { s += __shfl_xor(s, o, 32); s2 += __shfl_xor(s2, o, 32); }
    int w = threadIdx.x >> 5, l = threadIdx.x & 31;
    if (l == 0) { red[0][w] = s; red[1][w] = s2; }
    __syncthreads();
    if (w == 0) {
        s  = (l < 8) ? red[0][l] : 0.f;
        s2 = (l < 8) ? red[1][l] : 0.f;
        for (int o = 4; o > 0; o >>= 1) { s += __shfl_xor(s, o, 32); s2 += __shfl_xor(s2, o, 32); }
        if (l == 0) { red[0][0] = s; red[1][0] = s2; }
    }
    __syncthreads();
    float mean = red[0][0] / Dd;
    float var  = red[1][0] / Dd - mean * mean;
    float r = rsqrtf(var + 1e-5f);
    for (int i = threadIdx.x; i < Dd; i += 256)
        out[(size_t)row * Dd + i] = (_Float16)((xr[i] - mean) * r * g[i] + b[i]);
}

// ---------------- generic WMMA GEMM (double-buffered LDS staging) ----------------
// C[M,N] = A[M,K] (f16, row-major, lda) @ Wt[N,K] (f16) + bias, epilogue per MODE.
// MODE 0: out f16 (QKV)                MODE 1: out f32 = resid + acc + bias (out-proj)
// MODE 2: gathered rows, GELU, out f16 (FFN1)   MODE 3: expert rows, out f32 (FFN2)
template <int MODE>
__global__ __launch_bounds__(256) void gemm_wmma(
    const _Float16* __restrict__ A, int lda,
    const _Float16* __restrict__ W,           // [N,K] (per-expert stride N*K for MODE>=2)
    const float* __restrict__ bias,           // [N]  (per-expert stride N for MODE>=2)
    float* __restrict__ Cf, _Float16* __restrict__ Ch, int ldc,
    const float* __restrict__ resid,
    const int* __restrict__ list, const int* __restrict__ counts, const int* __restrict__ offsets,
    int M, int Nn, int Kd) {

    int e = (MODE >= 2) ? blockIdx.z : 0;
    int rowBase = blockIdx.y * 64;
    int colBase = blockIdx.x * 256;
    int cnt = M, off = 0;
    if (MODE >= 2) {
        cnt = counts[e];
        off = offsets[e];
        if (rowBase >= cnt) return;          // uniform across block
        W    += (size_t)e * Nn * Kd;
        bias += (size_t)e * Nn;
    }

    __shared__ __align__(16) _Float16 smA[2][64 * 32];

    int tid  = threadIdx.x;
    int srow = tid >> 2;                     // 0..63
    int scol = (tid & 3) * 8;                // 0/8/16/24
    int gi   = rowBase + srow;
    int arow; bool svalid;
    if (MODE == 2)      { svalid = gi < cnt; arow = svalid ? list[(size_t)e * NT + gi] : 0; }
    else if (MODE == 3) { svalid = gi < cnt; arow = off + (svalid ? gi : 0); }
    else                { svalid = gi < M;   arow = svalid ? gi : 0; }
    const _Float16* aPtr = A + (size_t)arow * lda + scol;

    int wid = tid >> 5, lane = tid & 31;
    int wm = wid >> 2, wn = wid & 3;
    int lh = lane >> 4, l15 = lane & 15;

    // issue the staging of one 64x32 A-tile into buffer `buf` (no wait)
    auto stage_tile = [&](int buf, int k0s) {
        bool done = false;
#ifdef USE_TDM
        if (MODE != 2) {
            if (wid == 0) {
                const _Float16* tb = A + ((size_t)((MODE == 3) ? off : 0) + rowBase) * lda + k0s;
                unsigned remain = (unsigned)(((MODE >= 2) ? cnt : M) - rowBase);
                if (remain > 64u) remain = 64u;
                tdm_load_tile_f16(&smA[buf][0], tb, 32, 64, (unsigned)lda, remain,
                                  (unsigned long long)lda);
            }
            done = true;
        }
#endif
        if (!done) {
#ifdef USE_ASYNC_LDS
            async_cp16(aPtr + k0s, &smA[buf][srow * 32 + scol]);
#else
            *(uint4*)(&smA[buf][srow * 32 + scol]) = *(const uint4*)(aPtr + k0s);
#endif
        }
    };

    v8f acc[2][4] = {};

    stage_tile(0, 0);                        // prologue: tile 0 in flight
    for (int k0 = 0; k0 < Kd; k0 += 32) {
        int cur = (k0 >> 5) & 1;
        bool have_next = (k0 + 32) < Kd;
        if (have_next) stage_tile(cur ^ 1, k0 + 32);   // overlap next DMA with compute
        // wait for the *current* tile only (in-order completion per wave)
#ifdef USE_TDM
        if (MODE != 2) {
            if (wid == 0) {
                if (have_next) __builtin_amdgcn_s_wait_tensorcnt(1);
                else           __builtin_amdgcn_s_wait_tensorcnt(0);
            }
        } else
#endif
        {
#ifdef USE_ASYNC_LDS
            if (have_next) __builtin_amdgcn_s_wait_asynccnt(1);
            else           __builtin_amdgcn_s_wait_asynccnt(0);
#endif
        }
        __syncthreads();

        F16Frag af[2];
#pragma unroll
        for (int mt = 0; mt < 2; mt++) {
            int m = wm * 32 + mt * 16 + l15;
            const _Float16* lp = &smA[cur][m * 32 + lh * 8];
            af[mt].q[0] = *(const uint4*)lp;
            af[mt].q[1] = *(const uint4*)(lp + 16);
        }
#pragma unroll
        for (int nt = 0; nt < 4; nt++) {
            int n = colBase + wn * 64 + nt * 16 + l15;
            const _Float16* wp = W + (size_t)n * Kd + k0 + lh * 8;
            F16Frag bf;
            bf.q[0] = *(const uint4*)wp;
            bf.q[1] = *(const uint4*)(wp + 16);
            if (k0 + 32 < Kd) __builtin_prefetch(wp + 32, 0, 1);
#pragma unroll
            for (int mt = 0; mt < 2; mt++)
                acc[mt][nt] = wmma_f16(af[mt].v, bf.v, acc[mt][nt]);
        }
        __syncthreads();                     // buffer reuse fence
    }

#pragma unroll
    for (int mt = 0; mt < 2; mt++) {
#pragma unroll
        for (int nt = 0; nt < 4; nt++) {
            int n = colBase + wn * 64 + nt * 16 + l15;
            float bv = bias[n];
#pragma unroll
            for (int r = 0; r < 8; r++) {
                int i = rowBase + wm * 32 + mt * 16 + r + lh * 8;
                bool valid = (MODE >= 2) ? (i < cnt) : (i < M);
                if (!valid) continue;
                size_t crow = (MODE >= 2) ? (size_t)(off + i) : (size_t)i;
                float v = acc[mt][nt][r] + bv;
                if (MODE == 0) {
                    Ch[crow * ldc + n] = (_Float16)v;
                } else if (MODE == 1) {
                    Cf[crow * ldc + n] = resid[crow * ldc + n] + v;
                } else if (MODE == 2) {
                    float gv = 0.5f * v * (1.f + tanhf(0.7978845608028654f * (v + 0.044715f * v * v * v)));
                    Ch[crow * ldc + n] = (_Float16)gv;
                } else {
                    Cf[crow * ldc + n] = v;
                }
            }
        }
    }
}

// ---------------- flash attention: block = 4 waves sharing one (b,h) ----------------
// qkv: [NT, 3*Dd] f16 (Q | K | V), o: [NT, Dd] f16
// Double-buffered LDS: K tile (32 keys x 64 hd) + transposed V tile (Vt[hd][key]).
__global__ __launch_bounds__(128) void flash_attn_kernel(
    const _Float16* __restrict__ qkv,
    const float* __restrict__ rel_bias,      // [H, 2S-1]
    _Float16* __restrict__ o) {

    __shared__ __align__(16) _Float16 smK[2][32 * 64];
    __shared__ __align__(16) _Float16 smVt[2][64 * 32];

    int tid = threadIdx.x;
    int wid = tid >> 5, lane = tid & 31;
    int qtile = blockIdx.x * 4 + wid;
    int h = blockIdx.y, b = blockIdx.z;
    int l15 = lane & 15, lh = lane >> 4;
    const int ld = 3 * Dd;

    size_t baseTok = (size_t)b * Ss * ld;
    const _Float16* Q  = qkv + baseTok + h * HDd;
    const _Float16* Km = qkv + baseTok + Dd + h * HDd;
    const _Float16* V  = qkv + baseTok + 2 * Dd + h * HDd;
    const float* rb = rel_bias + (size_t)h * (2 * Ss - 1);

    int qbase = qtile * 16;
    int q = qbase + l15;

    // stage K (async issue) + V transposed (sync) for key block kb0 into buffer `buf`
    auto stage_kv = [&](int buf, int kb0) {
#pragma unroll
        for (int cc = 0; cc < 2; cc++) {
            int c = tid + cc * 128;                  // 256 chunks of 8 halfs
            int row = c >> 3, col8 = (c & 7) * 8;    // row = key, col8 = hd
            const _Float16* gp = Km + (size_t)(kb0 + row) * ld + col8;
#ifdef USE_ASYNC_LDS
            async_cp16(gp, &smK[buf][row * 64 + col8]);
#else
            *(uint4*)(&smK[buf][row * 64 + col8]) = *(const uint4*)gp;
#endif
        }
#pragma unroll
        for (int cc = 0; cc < 2; cc++) {
            int c = tid + cc * 128;
            int row = c >> 3, col8 = (c & 7) * 8;
            union { uint4 qv; _Float16 hv[8]; } u;
            u.qv = *(const uint4*)(V + (size_t)(kb0 + row) * ld + col8);
#pragma unroll
            for (int j = 0; j < 8; j++) smVt[buf][(col8 + j) * 32 + row] = u.hv[j];
        }
    };

    // Q^T as B-fragments (2 chunks of hd 32); lane = query column, halves = hd 8h..8h+7 and 16+8h..16+8h+7
    F16Frag bq[2];
#pragma unroll
    for (int c = 0; c < 2; c++) {
        const _Float16* p = Q + (size_t)q * ld + c * 32 + lh * 8;
        bq[c].q[0] = *(const uint4*)p;
        bq[c].q[1] = *(const uint4*)(p + 16);
    }

    v8f oacc[4] = {};
    float m = -INFINITY, lsum = 0.f;

    stage_kv(0, 0);                          // prologue
    for (int kb = 0; kb < Ss; kb += 32) {
        int cur = (kb >> 5) & 1;
        bool have_next = (kb + 32) < Ss;
        if (have_next) stage_kv(cur ^ 1, kb + 32);     // overlap next K/V DMA with compute
#ifdef USE_ASYNC_LDS
        if (have_next) __builtin_amdgcn_s_wait_asynccnt(2);   // 2 issues per stage batch
        else           __builtin_amdgcn_s_wait_asynccnt(0);
#endif
        __syncthreads();

        float pv[2][8];
        float tsv[2];
        float msub[2];
        float mold = m, mcur = m;
#pragma unroll
        for (int sub = 0; sub < 2; sub++) {
            // K-rows A-fragments from LDS
            F16Frag ak[2];
#pragma unroll
            for (int c = 0; c < 2; c++) {
                const _Float16* lp = &smK[cur][(sub * 16 + l15) * 64 + c * 32 + lh * 8];
                ak[c].q[0] = *(const uint4*)lp;
                ak[c].q[1] = *(const uint4*)(lp + 16);
            }
            // S^T tile: rows = keys, cols = queries
            v8f c8 = {};
            c8 = wmma_f16(ak[0].v, bq[0].v, c8);
            c8 = wmma_f16(ak[1].v, bq[1].v, c8);

            int ks = kb + sub * 16;
            float tmax = -INFINITY;
            float sc[8];
#pragma unroll
            for (int r = 0; r < 8; r++) {
                int key = ks + lh * 8 + r;
                float v = c8[r] * 0.125f + rb[q - key + Ss - 1];
                sc[r] = v;
                tmax = fmaxf(tmax, v);
            }
            tmax = fmaxf(tmax, __shfl_xor(tmax, 16, 32));
            float mn = fmaxf(mcur, tmax);
            float ts = 0.f;
#pragma unroll
            for (int r = 0; r < 8; r++) { pv[sub][r] = expf(sc[r] - mn); ts += pv[sub][r]; }
            ts += __shfl_xor(ts, 16, 32);
            tsv[sub] = ts;
            msub[sub] = mn;
            mcur = mn;
        }
        float coef0 = expf(msub[0] - msub[1]);      // rescale sub0 probs to final max
        float alpha = expf(mold - msub[1]);         // rescale running state
        lsum = lsum * alpha + tsv[0] * coef0 + tsv[1];
        m = msub[1];

        // rescale O accumulators (rows = queries -> redistribute per-query alpha)
#pragma unroll
        for (int r = 0; r < 8; r++) {
            float a_r = __shfl(alpha, r + 8 * lh, 16);
#pragma unroll
            for (int nt = 0; nt < 4; nt++) oacc[nt][r] *= a_r;
        }

        // pack P (16q x 32k) A-fragment straight from the two S^T C-fragments
        F16Frag pa;
#pragma unroll
        for (int r = 0; r < 8; r++) {
            pa.h[r]     = (_Float16)(pv[0][r] * coef0);
            pa.h[8 + r] = (_Float16)pv[1][r];
        }
        // V B-fragments from transposed LDS tile: keys 8h..8h+7 and 16+8h..16+8h+7 per lane
#pragma unroll
        for (int nt = 0; nt < 4; nt++) {
            const _Float16* vp = &smVt[cur][(nt * 16 + l15) * 32 + lh * 8];
            F16Frag vb;
            vb.q[0] = *(const uint4*)vp;
            vb.q[1] = *(const uint4*)(vp + 16);
            oacc[nt] = wmma_f16(pa.v, vb.v, oacc[nt]);
        }
        __syncthreads();                     // buffer reuse fence
    }

    float linv = 1.f / lsum;
#pragma unroll
    for (int r = 0; r < 8; r++) {
        float li = __shfl(linv, r + 8 * lh, 16);
        int qq = qbase + r + 8 * lh;
        size_t orow = ((size_t)b * Ss + qq) * Dd + h * HDd;
#pragma unroll
        for (int nt = 0; nt < 4; nt++)
            o[orow + nt * 16 + l15] = (_Float16)(oacc[nt][r] * li);
    }
}

// ---------------- caption projection: cap[b,d] = text@cap_w + cap_b ----------------
__global__ void cap_proj_kernel(const float* __restrict__ ts, const float* __restrict__ w,
                                const float* __restrict__ bias, float* __restrict__ out) {
    int idx = blockIdx.x * 256 + threadIdx.x;       // B*D
    int b = idx / Dd, d = idx % Dd;
    float acc = bias[d];
    const float* t = ts + (size_t)b * Dd;
    for (int k = 0; k < Dd; k++) acc += t[k] * w[(size_t)k * Dd + d];
    out[idx] = acc;
}

// capgate[b,e] = cap[b] @ gate_w + gate_b
__global__ void cap_gate_kernel(const float* __restrict__ cap, const float* __restrict__ gw,
                                const float* __restrict__ gb, float* __restrict__ out) {
    int idx = threadIdx.x;
    if (idx < Bb * Ee) {
        int b = idx / Ee, e = idx % Ee;
        float acc = gb[e];
        for (int d = 0; d < Dd; d++) acc += cap[(size_t)b * Dd + d] * gw[(size_t)d * Ee + e];
        out[idx] = acc;
    }
}

// logits[t,e] = h2[t] @ gate_w[:,e] + capgate[b,e]
__global__ __launch_bounds__(256) void gate_logits_kernel(const _Float16* __restrict__ h2,
                                                          const float* __restrict__ gw,
                                                          const float* __restrict__ capgate,
                                                          float* __restrict__ logits) {
    int t = blockIdx.x;
    int e = threadIdx.x >> 5, l = threadIdx.x & 31;
    const _Float16* hr = h2 + (size_t)t * Dd;
    float acc = 0.f;
    for (int d = l; d < Dd; d += 32) acc += (float)hr[d] * gw[(size_t)d * Ee + e];
    for (int o = 16; o > 0; o >>= 1) acc += __shfl_xor(acc, o, 32);
    if (l == 0) logits[(size_t)t * Ee + e] = acc + capgate[(t / Ss) * Ee + e];
}

__global__ void init_small_kernel(int* counts, float* load) {
    int i = threadIdx.x;
    if (i < Ee) { counts[i] = 0; load[i] = 0.f; }
}

__global__ void topk_gate_kernel(const float* __restrict__ logits, int* __restrict__ counts,
                                 int* __restrict__ list, int* __restrict__ e_arr,
                                 int* __restrict__ i_arr, float* __restrict__ g_arr,
                                 float* __restrict__ load) {
    int t = blockIdx.x * 256 + threadIdx.x;
    if (t >= NT) return;
    const float* lg = logits + (size_t)t * Ee;
    int e0 = 0; float v0 = lg[0];
    for (int e = 1; e < Ee; e++) if (lg[e] > v0) { v0 = lg[e]; e0 = e; }
    int e1 = (e0 == 0) ? 1 : 0; float v1 = -INFINITY;
    for (int e = 0; e < Ee; e++) { if (e == e0) continue; if (lg[e] > v1) { v1 = lg[e]; e1 = e; } }
    float ex = expf(v1 - v0);
    float g0 = 1.f / (1.f + ex);
    float g1 = ex / (1.f + ex);
    int i0 = atomicAdd(&counts[e0], 1); list[(size_t)e0 * NT + i0] = t;
    int i1 = atomicAdd(&counts[e1], 1); list[(size_t)e1 * NT + i1] = t;
    e_arr[t * 2] = e0; i_arr[t * 2] = i0; g_arr[t * 2] = g0;
    e_arr[t * 2 + 1] = e1; i_arr[t * 2 + 1] = i1; g_arr[t * 2 + 1] = g1;
    const float inc = 1.f / (float)(NT * Kt);
    atomicAdd(&load[e0], inc);
    atomicAdd(&load[e1], inc);
}

__global__ void scan_offsets_kernel(const int* counts, int* offsets) {
    if (threadIdx.x == 0) {
        int a = 0;
        for (int e = 0; e < Ee; e++) { offsets[e] = a; a += counts[e]; }
    }
}

// x_out = x2 + sum_k gw_k * Z[row_k]
__global__ void combine_kernel(const float* __restrict__ x2, const float* __restrict__ Z,
                               const int* __restrict__ e_arr, const int* __restrict__ i_arr,
                               const float* __restrict__ g_arr, const int* __restrict__ offsets,
                               float* __restrict__ out) {
    int idx = blockIdx.x * 256 + threadIdx.x;   // NT*Dd
    int t = idx / Dd, d = idx % Dd;
    float acc = x2[idx];
#pragma unroll
    for (int k = 0; k < Kt; k++) {
        int e = e_arr[t * 2 + k];
        int i = i_arr[t * 2 + k];
        float g = g_arr[t * 2 + k];
        acc += g * Z[(size_t)(offsets[e] + i) * Dd + d];
    }
    out[idx] = acc;
}

// ---------------- host launcher ----------------
extern "C" void kernel_launch(void* const* d_in, const int* in_sizes, int n_in,
                              void* d_out, int out_size, void* d_ws, size_t ws_size,
                              hipStream_t stream) {
    const float* x        = (const float*)d_in[0];
    const float* text     = (const float*)d_in[1];
    const float* ln1_g    = (const float*)d_in[2];
    const float* ln1_b    = (const float*)d_in[3];
    const float* wqkv     = (const float*)d_in[4];
    const float* bqkv     = (const float*)d_in[5];
    const float* wo       = (const float*)d_in[6];
    const float* bo       = (const float*)d_in[7];
    const float* rel_bias = (const float*)d_in[8];
    const float* ln2_g    = (const float*)d_in[9];
    const float* ln2_b    = (const float*)d_in[10];
    const float* cap_w    = (const float*)d_in[11];
    const float* cap_b    = (const float*)d_in[12];
    const float* gate_w   = (const float*)d_in[13];
    const float* gate_b   = (const float*)d_in[14];
    const float* w1       = (const float*)d_in[15];
    const float* b1       = (const float*)d_in[16];
    const float* w2       = (const float*)d_in[17];
    const float* b2       = (const float*)d_in[18];

    float* out_x    = (float*)d_out;
    float* out_load = (float*)d_out + (size_t)NT * Dd;

    // workspace carve-up
    char* base = (char*)d_ws;
    size_t cur = 0;
    auto take = [&](size_t bytes) -> void* {
        void* r = base + cur;
        cur = (cur + bytes + 255) & ~(size_t)255;
        return r;
    };
    _Float16* h1_f16   = (_Float16*)take((size_t)NT * Dd * 2);
    _Float16* qkv_f16  = (_Float16*)take((size_t)NT * 3 * Dd * 2);
    _Float16* o_f16    = (_Float16*)take((size_t)NT * Dd * 2);
    float*    x2_f32   = (float*)take((size_t)NT * Dd * 4);
    _Float16* h2_f16   = (_Float16*)take((size_t)NT * Dd * 2);
    float*    cap_f32  = (float*)take((size_t)Bb * Dd * 4);
    float*    capg_f32 = (float*)take((size_t)Bb * Ee * 4);
    float*    logits   = (float*)take((size_t)NT * Ee * 4);
    _Float16* wqkv_t   = (_Float16*)take((size_t)3 * Dd * Dd * 2);
    _Float16* wo_t     = (_Float16*)take((size_t)Dd * Dd * 2);
    _Float16* w1_t     = (_Float16*)take((size_t)Ee * HIDd * Dd * 2);
    _Float16* w2_t     = (_Float16*)take((size_t)Ee * Dd * HIDd * 2);
    int*      counts   = (int*)take(Ee * 4);
    int*      offsets  = (int*)take(Ee * 4);
    int*      list     = (int*)take((size_t)Ee * NT * 4);
    int*      e_arr    = (int*)take((size_t)NT * Kt * 4);
    int*      i_arr    = (int*)take((size_t)NT * Kt * 4);
    float*    g_arr    = (float*)take((size_t)NT * Kt * 4);
    _Float16* Y_f16    = (_Float16*)take((size_t)NAS * HIDd * 2);
    float*    Z_f32    = (float*)take((size_t)NAS * Dd * 4);
    (void)ws_size; (void)in_sizes; (void)n_in; (void)out_size;

    // 1) weight conversion (f32 -> f16 transposed [N,K])
    cvt_transpose_kernel<<<dim3((Dd * 3 * Dd + 255) / 256, 1, 1), 256, 0, stream>>>(wqkv, wqkv_t, Dd, 3 * Dd);
    cvt_transpose_kernel<<<dim3((Dd * Dd + 255) / 256, 1, 1), 256, 0, stream>>>(wo, wo_t, Dd, Dd);
    cvt_transpose_kernel<<<dim3((Dd * HIDd + 255) / 256, 1, Ee), 256, 0, stream>>>(w1, w1_t, Dd, HIDd);
    cvt_transpose_kernel<<<dim3((HIDd * Dd + 255) / 256, 1, Ee), 256, 0, stream>>>(w2, w2_t, HIDd, Dd);

    // 2) LN1 -> h1 (f16)
    ln_f16_kernel<<<NT, 256, 0, stream>>>(x, ln1_g, ln1_b, h1_f16);

    // 3) QKV GEMM: [NT,1024] @ [1024,3072] -> f16
    gemm_wmma<0><<<dim3(3 * Dd / 256, NT / 64, 1), 256, 0, stream>>>(
        h1_f16, Dd, wqkv_t, bqkv, nullptr, qkv_f16, 3 * Dd,
        nullptr, nullptr, nullptr, nullptr, NT, 3 * Dd, Dd);

    // 4) flash attention (WMMA, online softmax, rel bias, double-buffered K/V tiles)
    flash_attn_kernel<<<dim3(Ss / 64, Hh, Bb), 128, 0, stream>>>(qkv_f16, rel_bias, o_f16);

    // 5) out-proj + residual: x2 = x + o@wo + bo
    gemm_wmma<1><<<dim3(Dd / 256, NT / 64, 1), 256, 0, stream>>>(
        o_f16, Dd, wo_t, bo, x2_f32, nullptr, Dd,
        x, nullptr, nullptr, nullptr, NT, Dd, Dd);

    // 6) LN2 -> h2 (f16)
    ln_f16_kernel<<<NT, 256, 0, stream>>>(x2_f32, ln2_g, ln2_b, h2_f16);

    // 7) gating path
    cap_proj_kernel<<<(Bb * Dd + 255) / 256, 256, 0, stream>>>(text, cap_w, cap_b, cap_f32);
    cap_gate_kernel<<<1, 32, 0, stream>>>(cap_f32, gate_w, gate_b, capg_f32);
    gate_logits_kernel<<<NT, 256, 0, stream>>>(h2_f16, gate_w, capg_f32, logits);
    init_small_kernel<<<1, 32, 0, stream>>>(counts, out_load);
    topk_gate_kernel<<<NT / 256, 256, 0, stream>>>(logits, counts, list, e_arr, i_arr, g_arr, out_load);
    scan_offsets_kernel<<<1, 32, 0, stream>>>(counts, offsets);

    // 8) expert FFN, only selected (K=2) experts: gathered WMMA GEMMs
    gemm_wmma<2><<<dim3(HIDd / 256, NT / 64, Ee), 256, 0, stream>>>(
        h2_f16, Dd, w1_t, b1, nullptr, Y_f16, HIDd,
        nullptr, list, counts, offsets, NT, HIDd, Dd);
    gemm_wmma<3><<<dim3(Dd / 256, NT / 64, Ee), 256, 0, stream>>>(
        Y_f16, HIDd, w2_t, b2, Z_f32, nullptr, Dd,
        nullptr, list, counts, offsets, NT, Dd, HIDd);

    // 9) combine: x_out = x2 + sum_k gw*Z
    combine_kernel<<<(NT * Dd) / 256, 256, 0, stream>>>(x2_f32, Z_f32, e_arr, i_arr, g_arr, offsets, out_x);
}
